// Attention_17952963297721
// MI455X (gfx1250) — matrix operand order repaired and checked by reading, not compiled, not run on hardware
//
#include <hip/hip_runtime.h>
#include <hip/hip_bf16.h>
#include <stdint.h>

// ---- CDNA5 WMMA vector types -------------------------------------------------
typedef __attribute__((ext_vector_type(16))) __bf16 v16bf;
typedef __attribute__((ext_vector_type(8)))  float  v8f;
typedef __attribute__((ext_vector_type(2)))  __bf16 bf2;
typedef __attribute__((ext_vector_type(4)))  __bf16 bf4;

#define B_N   4
#define S_LEN 4096
#define E_DIM 1024
#define H_DIM 128
#define WPB   4            /* waves per block in the attention kernel */
#define NEGINF (-1.0e30f)
#define SM_SCALE 0.08838834764831845f  /* 1/sqrt(128) */

static __device__ __forceinline__ v8f wmma_bf16(v16bf a, v16bf b, v8f c) {
  return __builtin_amdgcn_wmma_f32_16x16x32_bf16(false, a, false, b, (short)0, c,
                                                 false, false);
}

// ---- Kernel 0a: fp32 -> bf16 bulk convert (embds) ---------------------------
__global__ __launch_bounds__(256) void cvt_bf16_kernel(const float* __restrict__ src,
                                                       __bf16* __restrict__ dst,
                                                       int n4) {
  int i = blockIdx.x * blockDim.x + threadIdx.x;
  if (i < n4) {
    float4 f = ((const float4*)src)[i];
    bf4 o;
    o.x = (__bf16)f.x; o.y = (__bf16)f.y; o.z = (__bf16)f.z; o.w = (__bf16)f.w;
    ((bf4*)dst)[i] = o;
  }
}

// ---- Kernel 0b: W[e][h] -> Wt[h][e] bf16 (contiguous B-fragments) -----------
__global__ __launch_bounds__(256) void w_transpose_kernel(const float* __restrict__ W,
                                                          __bf16* __restrict__ Wt) {
  int i = blockIdx.x * blockDim.x + threadIdx.x;   // over H*E
  if (i < E_DIM * H_DIM) {
    int h = i / E_DIM, e = i - h * E_DIM;
    Wt[i] = (__bf16)W[(size_t)e * H_DIM + h];
  }
}

// ---- Kernel 1: projection  out[row][h] = embB[row][:] @ Wt[h][:]^T + bias ---
// One wave computes a 16-row x 128-col strip. Ping-pong software pipeline:
// two buffer sets alternate roles every k-step (no rotation copies, no
// WMMA->VALU hazard NOPs); loads for step ks+1 fly during step ks's WMMAs.
__global__ __launch_bounds__(32) void qkv_proj_kernel(
    const __bf16* __restrict__ embB, const __bf16* __restrict__ Wt,
    const float* __restrict__ bias, __bf16* __restrict__ dstRM,
    __bf16* __restrict__ dstTP, int transposeOut) {
  const int lane    = threadIdx.x & 31;
  const int n       = lane & 15;   // A: row M=n ; B/D: col N=n
  const int halfSel = lane >> 4;   // K-half for A/B, row-half for D

  const int rowBase = blockIdx.x * 16;
  const __bf16* arow = embB + (size_t)(rowBase + n) * E_DIM;
  const __bf16* wrow = Wt + (size_t)n * E_DIM + halfSel * 16;

  auto loadA = [&](int ks, v16bf& a) {
#pragma unroll
    for (int v = 0; v < 8; ++v) {
      int k = ks * 32 + halfSel * 16 + v * 2;
      bf2 p = *(const bf2*)(arow + k);
      a[2 * v] = p.x; a[2 * v + 1] = p.y;
    }
  };
  auto loadB = [&](int ks, v16bf* bb) {
#pragma unroll
    for (int hb = 0; hb < 8; ++hb)
      bb[hb] = *(const v16bf*)(wrow + (size_t)hb * 16 * E_DIM + ks * 32);
  };

  v8f acc[8] = {};
  v16bf A0, B0[8], A1, B1[8];
  const int NK = E_DIM / 32;  // 32 k-steps
  loadA(0, A0);
  loadB(0, B0);

  for (int ks = 0; ks < NK - 2; ks += 2) {
    loadA(ks + 1, A1);
    loadB(ks + 1, B1);
#pragma unroll
    for (int hb = 0; hb < 8; ++hb) acc[hb] = wmma_bf16(A0, B0[hb], acc[hb]);
    loadA(ks + 2, A0);
    loadB(ks + 2, B0);
#pragma unroll
    for (int hb = 0; hb < 8; ++hb) acc[hb] = wmma_bf16(A1, B1[hb], acc[hb]);
  }
  // tail: k-steps NK-2, NK-1
  loadA(NK - 1, A1);
  loadB(NK - 1, B1);
#pragma unroll
  for (int hb = 0; hb < 8; ++hb) acc[hb] = wmma_bf16(A0, B0[hb], acc[hb]);
#pragma unroll
  for (int hb = 0; hb < 8; ++hb) acc[hb] = wmma_bf16(A1, B1[hb], acc[hb]);

  // bias add + store
#pragma unroll
  for (int hb = 0; hb < 8; ++hb) {
    float bv = bias[hb * 16 + n];
#pragma unroll
    for (int r = 0; r < 8; ++r) {
      float val = acc[hb][r] + bv;
      int M   = r + 8 * halfSel;
      int row = rowBase + M;
      if (!transposeOut) {
        dstRM[(size_t)row * H_DIM + hb * 16 + n] = (__bf16)val;
      } else {
        int b = row / S_LEN, s = row - b * S_LEN;
        dstTP[((size_t)b * H_DIM + hb * 16 + n) * S_LEN + s] = (__bf16)val;
      }
    }
  }
}

// ---- Kernel 2: causal flash attention, 4 waves split the key dimension -----
// Block = 128 threads = 4 waves on one 16-query tile. Wave w handles chunks
// c = w, w+4, ... with private online-softmax state; partials merged in LDS.
// Ping-pong pipeline: next chunk's K fragments + this chunk's V fragments are
// issued right after the QK^T WMMAs; softmax VALU hides their latency; the
// two K buffer sets alternate roles (no rotation copies).
__global__ __launch_bounds__(128) void flash_attn_kernel(
    const __bf16* __restrict__ Qb, const __bf16* __restrict__ Kb,
    const __bf16* __restrict__ Vt, float* __restrict__ out) {
  const int lane    = threadIdx.x & 31;
  const int wave    = threadIdx.x >> 5;
  const int n       = lane & 15;
  const int halfSel = lane >> 4;
  const int qBase   = blockIdx.x * 16;
  const int b       = blockIdx.y;

  __shared__ __bf16 Pbuf[WPB][16 * 32];        // per-wave P staging (D->A layout)
  __shared__ float  OmLDS[WPB][16][H_DIM];     // per-wave partial O
  __shared__ float  mLDS[WPB][16], lLDS[WPB][16];

  // Q A-fragments: head dim 128 = 4 k-steps of 32 (kept in registers)
  v16bf Afrag[4];
  const __bf16* qrow = Qb + ((size_t)b * S_LEN + qBase + n) * H_DIM;
#pragma unroll
  for (int ks = 0; ks < 4; ++ks) {
#pragma unroll
    for (int v = 0; v < 8; ++v) {
      int k = ks * 32 + halfSel * 16 + v * 2;
      bf2 p = *(const bf2*)(qrow + k);
      Afrag[ks][2 * v] = p.x; Afrag[ks][2 * v + 1] = p.y;
    }
  }

  const __bf16* krow0 = Kb + ((size_t)b * S_LEN + n) * H_DIM + halfSel * 16;
  const __bf16* vrow0 = Vt + ((size_t)b * H_DIM + n) * S_LEN + halfSel * 16;

  auto loadBk = [&](int jB, v16bf* bk) {
#pragma unroll
    for (int t = 0; t < 2; ++t)
#pragma unroll
      for (int ks = 0; ks < 4; ++ks)
        bk[t * 4 + ks] =
            *(const v16bf*)(krow0 + (size_t)(jB + t * 16) * H_DIM + ks * 32);
  };
  auto loadBv = [&](int jB, v16bf* bv) {
#pragma unroll
    for (int hb = 0; hb < 8; ++hb)
      bv[hb] = *(const v16bf*)(vrow0 + (size_t)hb * 16 * S_LEN + jB);
  };

  v8f   Oacc[8] = {};
  float mrun[8], lrun[8];
#pragma unroll
  for (int r = 0; r < 8; ++r) { mrun[r] = NEGINF; lrun[r] = 0.0f; }

  const int nChunks = qBase / 32 + 1;  // causal: keys [0, qBase+16)

  // one chunk of 32 keys; BkCur holds this chunk's K frags, prefetch into BkNext
  auto chunkBody = [&](int c, v16bf* BkCur, v16bf* BkNext) {
    const int jBase = c * 32;

    // ---- S = Q @ K^T (K fragments loaded by previous stage / prolog)
    v8f sAcc[2] = {};
#pragma unroll
    for (int t = 0; t < 2; ++t)
#pragma unroll
      for (int ks = 0; ks < 4; ++ks)
        sAcc[t] = wmma_bf16(Afrag[ks], BkCur[t * 4 + ks], sAcc[t]);

    // ---- prefetch: this chunk's V fragments + next chunk's K fragments
    v16bf Bv[8];
    loadBv(jBase, Bv);
    if (c + WPB < nChunks) loadBk((c + WPB) * 32, BkNext);  // wave-uniform

    // ---- scale + causal mask + online softmax (rows M = r + 8*halfSel)
    float sval[2][8];
#pragma unroll
    for (int t = 0; t < 2; ++t)
#pragma unroll
      for (int r = 0; r < 8; ++r) {
        float s  = sAcc[t][r] * SM_SCALE;
        int   qi = qBase + r + 8 * halfSel;
        int   ki = jBase + t * 16 + n;
        sval[t][r] = (ki > qi) ? NEGINF : s;
      }

#pragma unroll
    for (int r = 0; r < 8; ++r) {
      float mx = fmaxf(sval[0][r], sval[1][r]);
#pragma unroll
      for (int mk = 8; mk >= 1; mk >>= 1) mx = fmaxf(mx, __shfl_xor(mx, mk, 32));
      float mnew  = fmaxf(mrun[r], mx);
      float alpha = __expf(mrun[r] - mnew);
      float p0 = __expf(sval[0][r] - mnew);
      float p1 = __expf(sval[1][r] - mnew);
      float rs = p0 + p1;
#pragma unroll
      for (int mk = 8; mk >= 1; mk >>= 1) rs += __shfl_xor(rs, mk, 32);
      lrun[r] = lrun[r] * alpha + rs;
      mrun[r] = mnew;
#pragma unroll
      for (int hb = 0; hb < 8; ++hb) Oacc[hb][r] = Oacc[hb][r] * alpha;
      Pbuf[wave][(r + 8 * halfSel) * 32 + n]      = (__bf16)p0;
      Pbuf[wave][(r + 8 * halfSel) * 32 + 16 + n] = (__bf16)p1;
    }

    // ---- reload P in A-layout from LDS (in-wave DS ordering; private slice)
    v16bf Pfrag;
    {
      const __bf16* prow = &Pbuf[wave][n * 32];
#pragma unroll
      for (int v = 0; v < 8; ++v) {
        int k = halfSel * 16 + v * 2;
        bf2 p = *(const bf2*)(prow + k);
        Pfrag[2 * v] = p.x; Pfrag[2 * v + 1] = p.y;
      }
    }

    // ---- O += P @ V
#pragma unroll
    for (int hb = 0; hb < 8; ++hb) Oacc[hb] = wmma_bf16(Pfrag, Bv[hb], Oacc[hb]);
  };

  v16bf BkA[8], BkB[8];
  if (wave < nChunks) loadBk(wave * 32, BkA);  // wave-uniform prolog

  for (int c = wave; c < nChunks; c += 2 * WPB) {
    chunkBody(c, BkA, BkB);                       // prefetches c+WPB into BkB
    if (c + WPB < nChunks) chunkBody(c + WPB, BkB, BkA);  // prefetches into BkA
  }

  // ---- publish per-wave partials
#pragma unroll
  for (int hb = 0; hb < 8; ++hb)
#pragma unroll
    for (int r = 0; r < 8; ++r)
      OmLDS[wave][r + 8 * halfSel][hb * 16 + n] = Oacc[hb][r];
  if (n == 0) {
#pragma unroll
    for (int r = 0; r < 8; ++r) {
      mLDS[wave][r + 8 * halfSel] = mrun[r];
      lLDS[wave][r + 8 * halfSel] = lrun[r];
    }
  }
  __syncthreads();

  // ---- merge 4 partials (log-sum-exp) ; wave w finalizes h-blocks {2w, 2w+1}
  float* orow = out + ((size_t)b * S_LEN + qBase) * H_DIM;
#pragma unroll
  for (int r = 0; r < 8; ++r) {
    int M = r + 8 * halfSel;
    float mglob = mLDS[0][M];
#pragma unroll
    for (int w = 1; w < WPB; ++w) mglob = fmaxf(mglob, mLDS[w][M]);
    float wexp[WPB];
    float lsum = 0.0f;
#pragma unroll
    for (int w = 0; w < WPB; ++w) {
      wexp[w] = __expf(mLDS[w][M] - mglob);
      lsum += wexp[w] * lLDS[w][M];
    }
    float inv = 1.0f / lsum;
#pragma unroll
    for (int j = 0; j < 2; ++j) {
      int hb = wave * 2 + j;
      float o = 0.0f;
#pragma unroll
      for (int w = 0; w < WPB; ++w) o += wexp[w] * OmLDS[w][M][hb * 16 + n];
      orow[(size_t)M * H_DIM + hb * 16 + n] = o * inv;
    }
  }
}

// ---- Host launcher ----------------------------------------------------------
extern "C" void kernel_launch(void* const* d_in, const int* in_sizes, int n_in,
                              void* d_out, int out_size, void* d_ws, size_t ws_size,
                              hipStream_t stream) {
  const float* embds = (const float*)d_in[0];
  const float* Wq = (const float*)d_in[1];
  const float* bq = (const float*)d_in[2];
  const float* Wk = (const float*)d_in[3];
  const float* bk = (const float*)d_in[4];
  const float* Wv = (const float*)d_in[5];
  const float* bv = (const float*)d_in[6];
  float* out = (float*)d_out;

  const size_t BS  = (size_t)B_N * S_LEN;
  const size_t BSE = BS * E_DIM;
  const size_t BSH = BS * H_DIM;
  const size_t EH  = (size_t)E_DIM * H_DIM;

  // workspace carve-up (bf16 buffers), ~45 MB total
  char* ws = (char*)d_ws;
  __bf16* embB = (__bf16*)ws;                 ws += BSE * 2;
  __bf16* WtQ  = (__bf16*)ws;                 ws += EH * 2;
  __bf16* WtK  = (__bf16*)ws;                 ws += EH * 2;
  __bf16* WtV  = (__bf16*)ws;                 ws += EH * 2;
  __bf16* Qb   = (__bf16*)ws;                 ws += BSH * 2;
  __bf16* Kb   = (__bf16*)ws;                 ws += BSH * 2;
  __bf16* Vt   = (__bf16*)ws;                 ws += BSH * 2;

  // 0a: embds -> bf16
  int n4 = (int)(BSE / 4);
  cvt_bf16_kernel<<<(n4 + 255) / 256, 256, 0, stream>>>(embds, embB, n4);

  // 0b: transpose+convert weights
  int nt = (int)EH;
  w_transpose_kernel<<<(nt + 255) / 256, 256, 0, stream>>>(Wq, WtQ);
  w_transpose_kernel<<<(nt + 255) / 256, 256, 0, stream>>>(Wk, WtK);
  w_transpose_kernel<<<(nt + 255) / 256, 256, 0, stream>>>(Wv, WtV);

  // 1: QKV projections (one wave per 16-row strip, ping-pong pipelined)
  dim3 pgrid((unsigned)(BS / 16));
  qkv_proj_kernel<<<pgrid, 32, 0, stream>>>(embB, WtQ, bq, Qb, nullptr, 0);
  qkv_proj_kernel<<<pgrid, 32, 0, stream>>>(embB, WtK, bk, Kb, nullptr, 0);
  qkv_proj_kernel<<<pgrid, 32, 0, stream>>>(embB, WtV, bv, nullptr, Vt, 1);

  // 2: causal flash attention (4 waves per q-tile, key-split + LDS merge)
  dim3 agrid(S_LEN / 16, B_N);
  flash_attn_kernel<<<agrid, 128, 0, stream>>>(Qb, Kb, Vt, out);

  (void)in_sizes; (void)n_in; (void)out_size; (void)ws_size;
}
